// GraphModule_91130616086924
// MI455X (gfx1250) — compile-verified, matrix operand order checked by reading, not run on hardware
//
#include <hip/hip_runtime.h>

// ---------------------------------------------------------------------------
// Types for CDNA5 WMMA (gfx1250, wave32)
// ---------------------------------------------------------------------------
typedef __attribute__((ext_vector_type(16))) _Float16 v16h;
typedef __attribute__((ext_vector_type(8)))  _Float16 v8h;
typedef __attribute__((ext_vector_type(8)))  float    v8f;

union Frag16 {
    v16h v;
    v8h  h2[2];
    _Float16 e[16];
};

#define BATCH 16
#define NN    1024   // H*W nodes
#define CC    256    // channels

// ---------------------------------------------------------------------------
// Row-major 16x32 f16 fragment loader matching the CDNA5 A-operand layout:
//   lanes 0-15 : row = lane,    K chunks [k..k+7]  and [k+16..k+23]
//   lanes 16-31: row = lane-16, K chunks [k+8..15] and [k+24..31]
// The B operand uses the same packing with "rows" = columns of B (i.e. rows
// of W for X @ W^T GEMMs).
// ---------------------------------------------------------------------------
__device__ __forceinline__ v16h load_frag(const _Float16* __restrict__ base,
                                          int row0, int stride, int k, int lane) {
    int r = row0 + (lane & 15);
    int h = lane >> 4;
    const _Float16* p = base + (long)r * stride + k + h * 8;
    Frag16 f;
    f.h2[0] = *(const v8h*)(p);
    f.h2[1] = *(const v8h*)(p + 16);
    return f.v;
}

__device__ __forceinline__ v8f wmma_f16(v16h a, v16h b, v8f c) {
    return __builtin_amdgcn_wmma_f32_16x16x32_f16(false, a, false, b,
                                                  (short)0, c, false, false);
}

// ---------------------------------------------------------------------------
// f32 -> f16 elementwise convert
// ---------------------------------------------------------------------------
__global__ void k_convert(const float* __restrict__ src,
                          _Float16* __restrict__ dst, int n) {
    int i = blockIdx.x * 256 + threadIdx.x;
    if (i < n) dst[i] = (_Float16)src[i];
}

// ---------------------------------------------------------------------------
// nodes_f16[b, n, c] = (f16) x[b, c, n]   (LDS-tiled 32x32 transpose)
// grid: BATCH * (CC/32) * (NN/32) blocks, blockDim (32, 8)
// ---------------------------------------------------------------------------
__global__ void k_transpose(const float* __restrict__ x,
                            _Float16* __restrict__ nodes) {
    __shared__ float tile[32][33];
    int b   = blockIdx.x >> 8;
    int rem = blockIdx.x & 255;
    int cT  = rem >> 5;   // 8 channel tiles
    int nT  = rem & 31;   // 32 node tiles
    int tx = threadIdx.x, ty = threadIdx.y;
#pragma unroll
    for (int i = 0; i < 4; ++i) {
        int c = cT * 32 + ty + 8 * i;
        int n = nT * 32 + tx;
        tile[ty + 8 * i][tx] = x[((long)b * CC + c) * NN + n];
    }
    __syncthreads();
#pragma unroll
    for (int i = 0; i < 4; ++i) {
        int n = nT * 32 + ty + 8 * i;
        int c = cT * 32 + tx;
        nodes[((long)b * NN + n) * CC + c] = (_Float16)tile[tx][ty + 8 * i];
    }
}

// ---------------------------------------------------------------------------
// Generic WMMA GEMM: out = epilogue(A @ W^T + bias), A:[B,NN,CC] f16,
// W:[CC,CC] f16.  One wave -> one 16x64 tile (A fragment reused across 4
// column tiles, 4 f32 accumulators), K = 256 -> 32 WMMAs per wave.
//   MODE 0: out_h = val                    (q projection)
//   MODE 1: out_h = gelu(val)              (MLP hidden)
//   MODE 2: out_f[b,c,n] = resid + val     (final, transposed residual add)
// Tiles: BATCH * (NN/16) * (CC/64) = 4096 waves = 512 blocks x 8 waves.
// ---------------------------------------------------------------------------
template <int MODE>
__global__ __launch_bounds__(256) void k_gemm(const _Float16* __restrict__ A,
                                              const _Float16* __restrict__ W,
                                              const float* __restrict__ bias,
                                              _Float16* __restrict__ outH,
                                              const float* __restrict__ resid,
                                              float* __restrict__ outF) {
    int wave = threadIdx.x >> 5;
    int lane = threadIdx.x & 31;
    long tile = (long)blockIdx.x * 8 + wave;       // 4096 tiles total
    int b     = (int)(tile >> 8);
    int t     = (int)(tile & 255);
    int mBase = (t >> 2) << 4;                     // 64 row tiles
    int nBase = (t & 3) << 6;                      // 4 groups of 64 cols
    const _Float16* Ab = A + (long)b * NN * CC;

    v8f acc[4] = {{}, {}, {}, {}};
#pragma unroll
    for (int kt = 0; kt < 8; ++kt) {
        v16h va = load_frag(Ab, mBase, CC, kt * 32, lane);
#pragma unroll
        for (int j = 0; j < 4; ++j) {
            v16h vb = load_frag(W, nBase + 16 * j, CC, kt * 32, lane);
            acc[j] = wmma_f16(va, vb, acc[j]);
        }
    }

    int h  = lane >> 4;
    int nc = lane & 15;

#pragma unroll
    for (int j = 0; j < 4; ++j) {
        int  nb   = nBase + 16 * j;
        float bcol = bias[nb + nc];
        if (MODE == 2) {
            // out[n, c] written transposed into y[b, c, n] + residual.
            int  cc   = nb + nc;
            long base = ((long)b * CC + cc) * NN + mBase + 8 * h;
            const float4* xp = (const float4*)(resid + base);
            float4* yp = (float4*)(outF + base);
            float4 r0 = xp[0], r1 = xp[1];
            r0.x += acc[j][0] + bcol; r0.y += acc[j][1] + bcol;
            r0.z += acc[j][2] + bcol; r0.w += acc[j][3] + bcol;
            r1.x += acc[j][4] + bcol; r1.y += acc[j][5] + bcol;
            r1.z += acc[j][6] + bcol; r1.w += acc[j][7] + bcol;
            yp[0] = r0; yp[1] = r1;
        } else {
#pragma unroll
            for (int v = 0; v < 8; ++v) {
                float val = acc[j][v] + bcol;
                if (MODE == 1)
                    val = 0.5f * val * (1.0f + erff(val * 0.70710678118654752f));
                outH[((long)b * NN + mBase + 8 * h + v) * CC + nb + nc] =
                    (_Float16)val;
            }
        }
    }
}

// ---------------------------------------------------------------------------
// Fused attention: one wave per (batch, 16-row query block).
//   scores strip 16x1024 (f32, /16 scale) -> LDS (64 KB)
//   per-row softmax (wave shuffles; 1/sum kept in the lane owning that row)
//   agg = P @ nodes via WMMA: kt outer, A-fragment built once per kt from
//   LDS f32 (normalized on the fly), 16 channel accumulators (128 VGPRs).
// ---------------------------------------------------------------------------
__global__ __launch_bounds__(32) void k_attn(const _Float16* __restrict__ q,
                                             const _Float16* __restrict__ xh,
                                             _Float16* __restrict__ agg) {
    __shared__ float sS[16 * NN];                  // 64 KB score strip
    int lane  = threadIdx.x;
    int b     = blockIdx.x >> 6;
    int rBase = (blockIdx.x & 63) << 4;
    const _Float16* qb = q + (long)b * NN * CC;
    int h  = lane >> 4;
    int nc = lane & 15;

    // Cache the 16x256 query block as 8 A-fragments (64 VGPRs).
    v16h qa[8];
#pragma unroll
    for (int kt = 0; kt < 8; ++kt)
        qa[kt] = load_frag(qb, rBase, CC, kt * 32, lane);

    // Phase 1: scores = (q . q^T) / 16 over all 64 column tiles.
    for (int j = 0; j < 64; ++j) {
        v8f acc = {};
#pragma unroll
        for (int kt = 0; kt < 8; ++kt) {
            v16h vb = load_frag(qb, j * 16, CC, kt * 32, lane);
            acc = wmma_f16(qa[kt], vb, acc);
        }
#pragma unroll
        for (int v = 0; v < 8; ++v)
            sS[(v + 8 * h) * NN + j * 16 + nc] = acc[v] * 0.0625f;
    }

    // Phase 2: per-row softmax; lane keeps 1/sum for the row it feeds (nc).
    float myInv = 0.0f;
    for (int r = 0; r < 16; ++r) {
        float* row = sS + r * NN;
        float m = -1e30f;
        for (int c = lane; c < NN; c += 32) m = fmaxf(m, row[c]);
#pragma unroll
        for (int off = 16; off >= 1; off >>= 1)
            m = fmaxf(m, __shfl_xor(m, off, 32));
        float s = 0.0f;
        for (int c = lane; c < NN; c += 32) {
            float e = expf(row[c] - m);
            row[c] = e;
            s += e;
        }
#pragma unroll
        for (int off = 16; off >= 1; off >>= 1)
            s += __shfl_xor(s, off, 32);
        if (nc == r) myInv = 1.0f / s;
    }

    // Phase 3: agg = P @ nodes (K = 1024), 16 channel-tile accumulators.
    const _Float16* xb = xh + (long)b * CC * NN;
    v8f acc[16];
#pragma unroll
    for (int ct = 0; ct < 16; ++ct) acc[ct] = (v8f){};

    for (int kt = 0; kt < 32; ++kt) {
        Frag16 fa;
        int base = nc * NN + kt * 32 + h * 8;
#pragma unroll
        for (int i = 0; i < 8; ++i) {
            fa.e[i]     = (_Float16)(sS[base + i]      * myInv);
            fa.e[8 + i] = (_Float16)(sS[base + 16 + i] * myInv);
        }
#pragma unroll
        for (int ct = 0; ct < 16; ++ct) {
            v16h vb = load_frag(xb, ct * 16, NN, kt * 32, lane);
            acc[ct] = wmma_f16(fa.v, vb, acc[ct]);
        }
    }

#pragma unroll
    for (int ct = 0; ct < 16; ++ct)
#pragma unroll
        for (int v = 0; v < 8; ++v)
            agg[((long)b * NN + rBase + 8 * h + v) * CC + ct * 16 + nc] =
                (_Float16)acc[ct][v];
}

// ---------------------------------------------------------------------------
// Launcher
// ---------------------------------------------------------------------------
extern "C" void kernel_launch(void* const* d_in, const int* in_sizes, int n_in,
                              void* d_out, int out_size, void* d_ws, size_t ws_size,
                              hipStream_t stream) {
    (void)in_sizes; (void)n_in; (void)out_size; (void)ws_size;
    const float* x  = (const float*)d_in[0];
    const float* pw = (const float*)d_in[1];
    const float* pb = (const float*)d_in[2];
    const float* w1 = (const float*)d_in[3];
    const float* b1 = (const float*)d_in[4];
    const float* w2 = (const float*)d_in[5];
    const float* b2 = (const float*)d_in[6];
    float* y = (float*)d_out;

    const long TENS = (long)BATCH * CC * NN;       // 4,194,304 elems
    _Float16* xh    = (_Float16*)d_ws;             // x as f16 (== nodes^T)
    _Float16* nodes = xh    + TENS;                // [B, N, C] f16
    _Float16* qh    = nodes + TENS;                // q         f16
    _Float16* aggh  = qh    + TENS;                // agg       f16
    _Float16* hh    = aggh  + TENS;                // MLP hid   f16
    _Float16* pwh   = hh    + TENS;                // weights   f16
    _Float16* w1h   = pwh + CC * CC;
    _Float16* w2h   = w1h + CC * CC;

    // 1) converts
    k_convert<<<(int)(TENS / 256), 256, 0, stream>>>(x,  xh,  (int)TENS);
    k_convert<<<CC * CC / 256,     256, 0, stream>>>(pw, pwh, CC * CC);
    k_convert<<<CC * CC / 256,     256, 0, stream>>>(w1, w1h, CC * CC);
    k_convert<<<CC * CC / 256,     256, 0, stream>>>(w2, w2h, CC * CC);

    // 2) transpose x -> nodes
    k_transpose<<<BATCH * 256, dim3(32, 8), 0, stream>>>(x, nodes);

    // 3) q = nodes @ pw^T + pb
    k_gemm<0><<<512, 256, 0, stream>>>(nodes, pwh, pb, qh, nullptr, nullptr);

    // 4) attention (scores -> softmax -> agg), one wave per 16 rows
    k_attn<<<BATCH * 64, 32, 0, stream>>>(qh, xh, aggh);

    // 5) h = gelu(agg @ w1^T + b1)
    k_gemm<1><<<512, 256, 0, stream>>>(aggh, w1h, b1, hh, nullptr, nullptr);

    // 6) y = x + (h @ w2^T + b2)^T
    k_gemm<2><<<512, 256, 0, stream>>>(hh, w2h, b2, nullptr, x, y);
}
